// EfficientSelfAttention_79293686219395
// MI455X (gfx1250) — matrix operand order repaired and checked by reading, not compiled, not run on hardware
//
#include <hip/hip_runtime.h>
#include <hip/hip_bf16.h>

typedef __attribute__((ext_vector_type(16))) _Float16 v16h;
typedef __attribute__((ext_vector_type(8)))  _Float16 v8h;
typedef __attribute__((ext_vector_type(4)))  _Float16 v4h;
typedef __attribute__((ext_vector_type(8)))  float    v8f;

union h16u { v16h v; v8h h[2]; };

#define WMMA_F16(a, b, c) \
    __builtin_amdgcn_wmma_f32_16x16x32_f16(false, (a), false, (b), (short)0, (c), false, false)

// Build a v16h fragment from two contiguous 16-byte runs (two b128 loads).
// Per ISA 7.12.2 each lane's 8 f16 for VGPRs 0-3 (K = khalf..khalf+7) and
// VGPRs 4-7 (K = 16+khalf..16+khalf+7) are consecutive in row-major memory.
__device__ __forceinline__ v16h frag_2x8(const _Float16* p0, const _Float16* p1) {
    h16u u;
    u.h[0] = *(const v8h*)p0;
    u.h[1] = *(const v8h*)p1;
    return u.v;
}

// ---------------------------------------------------------------------------
// One-time fp32 -> fp16 staging (4 elements/thread, packed cvt + b64 store).
// ---------------------------------------------------------------------------
__global__ void __launch_bounds__(256)
cvt_f32_f16(const float* __restrict__ in, _Float16* __restrict__ out, int n)
{
    const int i = (blockIdx.x * blockDim.x + threadIdx.x) * 4;
    if (i < n) {
        const float4 f = *(const float4*)(in + i);
        v4h o;
        o[0] = (_Float16)f.x; o[1] = (_Float16)f.y;
        o[2] = (_Float16)f.z; o[3] = (_Float16)f.w;
        *(v4h*)(out + i) = o;
    }
}

// ---------------------------------------------------------------------------
// One-time V transpose: Vt[b][h][d][m] = KV[b*256+m][256 + h*32 + d]
// Makes the P@V B-operand contiguous over its K dimension (Nr) -> b128 loads.
// ---------------------------------------------------------------------------
__global__ void __launch_bounds__(256)
vtrans_k(const _Float16* __restrict__ KV, _Float16* __restrict__ Vt)
{
    const int t = blockIdx.x * 256 + threadIdx.x;     // 262144 elements
    const int m = t & 255;
    const int d = (t >> 8) & 31;
    const int h = (t >> 13) & 7;
    const int b = t >> 16;
    Vt[(size_t)(((b * 8 + h) * 32) + d) * 256 + m] =
        KV[(size_t)(b * 256 + m) * 512 + 256 + h * 32 + d];
}

// ---------------------------------------------------------------------------
// SR=4 conv-as-GEMM A-operand gather (im2col without the buffer).
// ---------------------------------------------------------------------------
__device__ __forceinline__ int conv_addr(int m, int k) {
    const int b   = m >> 8;
    const int mm  = m & 255;
    const int pr  = mm >> 4;
    const int pc  = mm & 15;
    const int cin = k >> 4;
    const int r   = k & 15;
    const int i   = r >> 2;
    const int j   = r & 3;
    const int row = (pr * 4 + i) * 64 + (pc * 4 + j);
    return ((b << 12) + row) * 256 + cin;
}

// ---------------------------------------------------------------------------
// Wave-tile GEMM: C[M,N] = A[M,K] @ W[N,K]^T (+bias). A,W f16; f16 WMMA.
// Each wave computes 16x64 (1 A fragment re-used by 4 B fragments / K-step).
// MODE 0: row-major f16 A (2x b128 per fragment).  MODE 1: conv gather.
// OUT16 1: store f16 scaled by oscale.              OUT16 0: f32 + bias.
// ---------------------------------------------------------------------------
template <int MODE, int OUT16>
__global__ void __launch_bounds__(256)
gemm16(const _Float16* __restrict__ A, const _Float16* __restrict__ Wt,
       const float* __restrict__ bias, void* __restrict__ CoutV,
       int M, int N, int K, float oscale, const _Float16* __restrict__ Xsrc)
{
    const int lane   = threadIdx.x & 31;
    const int wave   = (blockIdx.x * blockDim.x + threadIdx.x) >> 5;
    const int ntiles = N >> 6;
    const int tm     = wave / ntiles;
    const int tn     = wave - tm * ntiles;
    if (tm * 16 >= M) return;

    const int m0    = tm * 16;
    const int n0    = tn * 64;
    const int laneM = lane & 15;
    const int hi    = lane >> 4;
    const int khalf = hi * 8;
    const int mrow  = m0 + laneM;

    v8f acc[4] = {};

    for (int kb = 0; kb < K; kb += 32) {
        v16h a;
        if (MODE == 0) {
            const _Float16* Arow = A + (size_t)mrow * K + kb;
            a = frag_2x8(Arow + khalf, Arow + 16 + khalf);
            if (kb + 32 < K)
                __builtin_prefetch(Arow + 32, 0, 1);   // global_prefetch_b8
        } else {
#pragma unroll
            for (int v = 0; v < 8; ++v) {
                const int kk = (v < 4) ? (khalf + 2 * v) : (16 + khalf + 2 * (v - 4));
                a[2 * v]     = Xsrc[conv_addr(mrow, kb + kk)];
                a[2 * v + 1] = Xsrc[conv_addr(mrow, kb + kk + 1)];
            }
        }
#pragma unroll
        for (int nb = 0; nb < 4; ++nb) {
            const int ncol = n0 + nb * 16 + laneM;               // N = lane%16
            const _Float16* wr = Wt + (size_t)ncol * K + kb;
            const v16h b = frag_2x8(wr + khalf, wr + 16 + khalf); // 2x16B runs
            acc[nb] = WMMA_F16(a, b, acc[nb]);
        }
    }

    // acc VGPR j -> row (j + hi*8), col = lane%16
    if (OUT16) {
        _Float16* C16 = (_Float16*)CoutV;
#pragma unroll
        for (int nb = 0; nb < 4; ++nb) {
            const int ncol = n0 + nb * 16 + laneM;
#pragma unroll
            for (int j = 0; j < 8; ++j)
                C16[(size_t)(m0 + hi * 8 + j) * N + ncol] = (_Float16)(acc[nb][j] * oscale);
        }
    } else {
        float* Cf = (float*)CoutV;
#pragma unroll
        for (int nb = 0; nb < 4; ++nb) {
            const int   ncol = n0 + nb * 16 + laneM;
            const float badd = bias ? bias[ncol] : 0.0f;
#pragma unroll
            for (int j = 0; j < 8; ++j)
                Cf[(size_t)(m0 + hi * 8 + j) * N + ncol] = acc[nb][j] + badd;
        }
    }
}

// ---------------------------------------------------------------------------
// LayerNorm over C=256, one block per row; f32 in, f16 out (staged).
// ---------------------------------------------------------------------------
__global__ void __launch_bounds__(256)
layernorm_k(const float* __restrict__ in, const float* __restrict__ gamma,
            const float* __restrict__ beta, _Float16* __restrict__ out)
{
    const int row = blockIdx.x;
    const int tid = threadIdx.x;
    const float v = in[row * 256 + tid];

    float a = v, b = v * v;
#pragma unroll
    for (int m = 16; m >= 1; m >>= 1) {
        a += __shfl_xor(a, m, 32);
        b += __shfl_xor(b, m, 32);
    }
    __shared__ float s1[8], s2[8];
    if ((tid & 31) == 0) { s1[tid >> 5] = a; s2[tid >> 5] = b; }
    __syncthreads();
    float ta = 0.f, tb = 0.f;
#pragma unroll
    for (int i = 0; i < 8; ++i) { ta += s1[i]; tb += s2[i]; }

    const float mean = ta * (1.0f / 256.0f);
    const float var  = tb * (1.0f / 256.0f) - mean * mean;
    const float r    = rsqrtf(var + 1e-5f);
    out[row * 256 + tid] = (_Float16)((v - mean) * r * gamma[tid] + beta[tid]);
}

// ---------------------------------------------------------------------------
// Flash-style attention, one wave per (b, head, 16-row q tile).
// Q is pre-scaled by d^-0.5 at its producing GEMM, so S needs no per-element
// scaling. Softmax normalization is deferred to the 16 output accumulators
// (linear), so P holds raw exp(). All fragments are 2x b128 loads.
// ---------------------------------------------------------------------------
__global__ void __launch_bounds__(128)
attn_k(const _Float16* __restrict__ Q, const _Float16* __restrict__ KV,
       const _Float16* __restrict__ Vt, _Float16* __restrict__ O)
{
    const int lane  = threadIdx.x & 31;
    const int wid   = threadIdx.x >> 5;
    const int wave  = blockIdx.x * 4 + wid;      // 8192 waves
    const int qtile = wave & 255;
    const int head  = (wave >> 8) & 7;
    const int b     = wave >> 11;

    const int laneM = lane & 15;
    const int hi    = lane >> 4;
    const int khalf = hi * 8;

    // Q A-fragment (16x32, pre-scaled): two b128 loads per lane
    const int m0 = qtile * 16;
    const _Float16* qr = Q + (size_t)(b * 4096 + m0 + laneM) * 256 + head * 32;
    const v16h aq = frag_2x8(qr + khalf, qr + 16 + khalf);

    // S = q @ k^T : 16 WMMAs, kept in 128 accumulator VGPRs
    const _Float16* kvb = KV + (size_t)(b * 256) * 512 + head * 32;
    v8f s[16];
#pragma unroll
    for (int blk = 0; blk < 16; ++blk) {
        const _Float16* kr = kvb + (size_t)(blk * 16 + laneM) * 512;
        const v16h bk = frag_2x8(kr + khalf, kr + 16 + khalf);
        v8f z = {};
        s[blk] = WMMA_F16(aq, bk, z);
    }

    // softmax stats over 256 cols: per-row max/sum with 16-lane butterflies
    float rinv[8];
#pragma unroll
    for (int j = 0; j < 8; ++j) {                // row = j + hi*8
        float m = -3.0e38f;
#pragma unroll
        for (int blk = 0; blk < 16; ++blk) m = fmaxf(m, s[blk][j]);
#pragma unroll
        for (int t = 8; t >= 1; t >>= 1) m = fmaxf(m, __shfl_xor(m, t, 32));
        float su = 0.f;
#pragma unroll
        for (int blk = 0; blk < 16; ++blk) {
            const float e = __expf(s[blk][j] - m);
            s[blk][j] = e;
            su += e;
        }
#pragma unroll
        for (int t = 8; t >= 1; t >>= 1) su += __shfl_xor(su, t, 32);
        rinv[j] = 1.0f / su;
    }

    // raw exp(P) -> LDS to re-swizzle into A-fragment layout
    __shared__ _Float16 Plds[4][16][264];        // 528B row stride (16B aligned)
    _Float16 (*P)[264] = Plds[wid];
#pragma unroll
    for (int blk = 0; blk < 16; ++blk)
#pragma unroll
        for (int j = 0; j < 8; ++j)
            P[hi * 8 + j][blk * 16 + laneM] = (_Float16)s[blk][j];
    __syncthreads();

    // out(16x32) = P(16x256) @ V(256x32); V read from Vt, contiguous over Nr
    const _Float16* vtb = Vt + (size_t)(b * 8 + head) * 32 * 256;
    v8f o0 = {}, o1 = {};
#pragma unroll
    for (int kc = 0; kc < 8; ++kc) {
        const int kb = kc * 32;
        const v16h ap = frag_2x8(&P[laneM][kb + khalf], &P[laneM][kb + 16 + khalf]);
#pragma unroll
        for (int nh = 0; nh < 2; ++nh) {
            const _Float16* vr = vtb + (size_t)(nh * 16 + laneM) * 256 + kb;
            const v16h bv = frag_2x8(vr + khalf, vr + 16 + khalf); // 2x16B runs over Nr
            if (nh == 0) o0 = WMMA_F16(ap, bv, o0);
            else         o1 = WMMA_F16(ap, bv, o1);
        }
    }

    // deferred softmax normalization + store f16 in (B, N, C) layout
    _Float16* ob = O + (size_t)(b * 4096 + m0) * 256 + head * 32;
#pragma unroll
    for (int j = 0; j < 8; ++j) {
        ob[(hi * 8 + j) * 256 + laneM]      = (_Float16)(o0[j] * rinv[j]);
        ob[(hi * 8 + j) * 256 + 16 + laneM] = (_Float16)(o1[j] * rinv[j]);
    }
}

// ---------------------------------------------------------------------------
// Host launch. ws layout (f16 unless noted), ~31 MB total:
// xh[4.19M] Wqh[64K] Wkvh[128K] srwh[1M] Wph[64K] Qh[4.19M] Xrh[256K]
// KVh[512K] Vth[256K] Oh[4.19M] Xpre(f32)[256K]
// ---------------------------------------------------------------------------
extern "C" void kernel_launch(void* const* d_in, const int* in_sizes, int n_in,
                              void* d_out, int out_size, void* d_ws, size_t ws_size,
                              hipStream_t stream) {
    const float* x     = (const float*)d_in[0];
    const float* Wq    = (const float*)d_in[1];
    const float* Wkv   = (const float*)d_in[2];
    const float* srw   = (const float*)d_in[3];   // (C,C,4,4) == (N=256, K=4096)
    const float* srb   = (const float*)d_in[4];
    const float* gamma = (const float*)d_in[5];
    const float* beta  = (const float*)d_in[6];
    const float* Wp    = (const float*)d_in[7];
    const float* bp    = (const float*)d_in[8];
    (void)in_sizes; (void)n_in; (void)out_size; (void)ws_size;

    _Float16* xh   = (_Float16*)d_ws;
    _Float16* Wqh  = xh   + 4194304;
    _Float16* Wkvh = Wqh  + 65536;
    _Float16* srwh = Wkvh + 131072;
    _Float16* Wph  = srwh + 1048576;
    _Float16* Qh   = Wph  + 65536;
    _Float16* Xrh  = Qh   + 4194304;
    _Float16* KVh  = Xrh  + 262144;
    _Float16* Vth  = KVh  + 524288;
    _Float16* Oh   = Vth  + 262144;
    float*    Xpre = (float*)(Oh + 4194304);
    float*    out  = (float*)d_out;

    const float qscale = 0.1767766952966369f;     // 32^-0.5, folded into Q

    // one-time fp16 staging of inputs/weights
    cvt_f32_f16<<<4096, 256, 0, stream>>>(x,   xh,   4194304);
    cvt_f32_f16<<<64,   256, 0, stream>>>(Wq,  Wqh,  65536);
    cvt_f32_f16<<<128,  256, 0, stream>>>(Wkv, Wkvh, 131072);
    cvt_f32_f16<<<1024, 256, 0, stream>>>(srw, srwh, 1048576);
    cvt_f32_f16<<<64,   256, 0, stream>>>(Wp,  Wph,  65536);

    // Q = (x @ Wq^T) * d^-0.5  (f16 out)  (M=16384, N=256, K=256)
    gemm16<0, 1><<<512, 256, 0, stream>>>(xh, Wqh, nullptr, Qh, 16384, 256, 256, qscale, nullptr);
    // SR conv as GEMM (fused gather)      (M=1024, N=256, K=4096), f32 out + bias
    gemm16<1, 0><<<32, 256, 0, stream>>>(nullptr, srwh, srb, Xpre, 1024, 256, 4096, 1.0f, xh);
    // LayerNorm (f32 in, f16 out)
    layernorm_k<<<1024, 256, 0, stream>>>(Xpre, gamma, beta, Xrh);
    // KV = Xr @ Wkv^T (f16 out)           (M=1024, N=512, K=256)
    gemm16<0, 1><<<64, 256, 0, stream>>>(Xrh, Wkvh, nullptr, KVh, 1024, 512, 256, 1.0f, nullptr);
    // V transpose for contiguous K-dim reads in P@V
    vtrans_k<<<1024, 256, 0, stream>>>(KVh, Vth);
    // Flash attention (f16 in/out): 8192 waves
    attn_k<<<2048, 128, 0, stream>>>(Qh, KVh, Vth, Oh);
    // out = O @ Wp^T + bp (f32 out)       (M=16384, N=256, K=256)
    gemm16<0, 0><<<512, 256, 0, stream>>>(Oh, Wph, bp, (void*)out, 16384, 256, 256, 1.0f, nullptr);
}
